// Module_60584808678061
// MI455X (gfx1250) — matrix operand order repaired and checked by reading, not compiled, not run on hardware
//
#include <hip/hip_runtime.h>
#include <hip/hip_bf16.h>
#include <math.h>

#define DIM    256
#define NKEY   128
#define NQ     4096
#define KROW   260   // padded floats per K row in LDS (breaks 256-stride bank conflicts)
#define WROW   264   // padded bf16 per W row in LDS

typedef __bf16 bf16_t;
typedef __attribute__((ext_vector_type(16))) bf16_t v16bf;
typedef __attribute__((ext_vector_type(8)))  float  v8f;

// ---- dynamic LDS layout (bytes) ----
#define OFF_KT   0
#define SZ_KT    (128*KROW*4)            // 133120
#define OFF_WT   (OFF_KT + SZ_KT)
#define SZ_WT    (256*WROW*2)            // 135168
#define OFF_SM   (OFF_WT + SZ_WT)        // 268288, 16B aligned
// float indices inside SM region
#define SM_WMU   0
#define SM_WLV   256
#define SM_BPR1  512
#define SM_WPR2  768
#define SM_GAM   1024
#define SM_BET   1280
#define SM_DMU   1536
#define SM_DLV   1664
#define SM_MPR   1792
#define SM_WTS   1920
#define SM_RED   2048
#define SM_FLOATS 2304
#define SMEM_BYTES (OFF_SM + SM_FLOATS*4)

__device__ __forceinline__ float blockSum(float* red, int t, float v) {
    red[t] = v; __syncthreads();
    #pragma unroll
    for (int off = 128; off > 0; off >>= 1) {
        if (t < off) red[t] += red[t + off];
        __syncthreads();
    }
    float r = red[0]; __syncthreads();
    return r;
}

__device__ __forceinline__ float blockMax(float* red, int t, float v) {
    red[t] = v; __syncthreads();
    #pragma unroll
    for (int off = 128; off > 0; off >>= 1) {
        if (t < off) red[t] = fmaxf(red[t], red[t + off]);
        __syncthreads();
    }
    float r = red[0]; __syncthreads();
    return r;
}

// ---------- kernel 1: W_pr1 fp32 -> bf16 (L2-resident operand for WMMA) ----------
__global__ __launch_bounds__(256) void convW_kernel(const float* __restrict__ W,
                                                    bf16_t* __restrict__ Wb) {
    int i = blockIdx.x * 256 + threadIdx.x;   // grid = 256 blocks -> 65536 elems
    Wb[i] = (bf16_t)W[i];
}

// ---------- kernel 2: fused variational attention, one workgroup per query ----------
__global__ __launch_bounds__(256)
void vattn_main(const float* __restrict__ Q,   const float* __restrict__ K,
                const float* __restrict__ V,   const float* __restrict__ eps,
                const float* __restrict__ w_mu_q, const float* __restrict__ w_mu_k,
                const float* __restrict__ b_mu,   const float* __restrict__ w_lv_q,
                const float* __restrict__ w_lv_k, const float* __restrict__ b_lv,
                const float* __restrict__ b_pr1,  const float* __restrict__ w_pr2,
                const float* __restrict__ b_pr2,  const float* __restrict__ gamma,
                const float* __restrict__ beta,   const bf16_t* __restrict__ Wbf,
                float* __restrict__ out, float* __restrict__ klPart) {
    extern __shared__ __align__(16) char smem[];
    float*  kT = (float*)(smem + OFF_KT);   // [128][KROW] fp32 K tile
    bf16_t* wT = (bf16_t*)(smem + OFF_WT);  // [256][WROW] bf16 W_pr1 (e-major, d contiguous)
    float*  sm = (float*)(smem + OFF_SM);

    const int t = threadIdx.x;
    const int q = blockIdx.x;

    // ---------- phase A: async-stage tiles into LDS (no VGPR round trip) ----------
    {
        // K[q]: 32768 floats, 16B per lane per op, GVS mode (SGPR base + u32 lane offset)
        const uint32_t kT_lds = (uint32_t)(uintptr_t)kT;
        const void* Kq = (const void*)(K + (size_t)q * (NKEY * DIM));
        #pragma unroll 4
        for (int i = 0; i < 32; ++i) {
            int f  = t + i * 256;          // float4 chunk index
            int kk = f >> 6, d4 = f & 63;
            uint32_t lds  = kT_lds + (uint32_t)(kk * KROW + d4 * 4) * 4u;
            uint32_t goff = (uint32_t)f * 16u;
            asm volatile("global_load_async_to_lds_b128 %0, %1, %2"
                         :: "v"(lds), "v"(goff), "s"(Kq) : "memory");
        }
        // W_pr1 bf16: 65536 bf16, 8 elems (16B) per lane per op
        const uint32_t wT_lds = (uint32_t)(uintptr_t)wT;
        const void* Wg = (const void*)Wbf;
        #pragma unroll 4
        for (int i = 0; i < 32; ++i) {
            int c = t + i * 256;           // 8-elem chunk index
            int e = c >> 5, d8 = c & 31;
            uint32_t lds  = wT_lds + (uint32_t)(e * WROW + d8 * 8) * 2u;
            uint32_t goff = (uint32_t)c * 16u;
            asm volatile("global_load_async_to_lds_b128 %0, %1, %2"
                         :: "v"(lds), "v"(goff), "s"(Wg) : "memory");
        }
        sm[SM_WMU  + t] = w_mu_k[t];
        sm[SM_WLV  + t] = w_lv_k[t];
        sm[SM_BPR1 + t] = b_pr1[t];
        sm[SM_WPR2 + t] = w_pr2[t];
        sm[SM_GAM  + t] = gamma[t];
        sm[SM_BET  + t] = beta[t];
    }
    // overlap the Q-dot global reads with the async transfers
    const float Qv = Q[(size_t)q * DIM + t];
    const float pm = Qv * w_mu_q[t];
    const float pl = Qv * w_lv_q[t];
    asm volatile("s_wait_asynccnt 0" ::: "memory");
    __syncthreads();

    float* red = sm + SM_RED;
    const float q_mu = blockSum(red, t, pm);
    const float q_lv = blockSum(red, t, pl);

    // ---------- phase B: posterior dot products (fp32) ----------
    {
        const int  row = t & 127;
        const float* kr = kT + row * KROW;
        const float* wv = sm + ((t < 128) ? SM_WMU : SM_WLV);
        float a0 = 0.f, a1 = 0.f, a2 = 0.f, a3 = 0.f;
        #pragma unroll 8
        for (int d = 0; d < 256; d += 4) {
            a0 = fmaf(kr[d    ], wv[d    ], a0);
            a1 = fmaf(kr[d + 1], wv[d + 1], a1);
            a2 = fmaf(kr[d + 2], wv[d + 2], a2);
            a3 = fmaf(kr[d + 3], wv[d + 3], a3);
        }
        sm[((t < 128) ? SM_DMU : SM_DLV) + row] = (a0 + a1) + (a2 + a3);
    }

    // ---------- phase C: prior GEMM with bf16 WMMA ----------
    {
        const int wv   = t >> 5;        // wave 0..7 -> M rows [16w, 16w+16)
        const int lane = t & 31;
        const int ml   = lane & 15;
        const int half = lane >> 4;
        const int m    = 16 * wv + ml;

        // preload the 8 A-fragments for this wave's M-tile row (fp32 LDS -> bf16 regs)
        v16bf afr[8];
        #pragma unroll
        for (int s = 0; s < 8; ++s) {
            const int dbase = 32 * s + 16 * half;
            const float4* rp = (const float4*)(kT + m * KROW);
            float4 f0 = rp[(dbase >> 2)    ];
            float4 f1 = rp[(dbase >> 2) + 1];
            float4 f2 = rp[((dbase + 8) >> 2)    ];
            float4 f3 = rp[((dbase + 8) >> 2) + 1];
            v16bf a;
            a[0]=(bf16_t)f0.x; a[1]=(bf16_t)f0.y; a[2]=(bf16_t)f0.z; a[3]=(bf16_t)f0.w;
            a[4]=(bf16_t)f1.x; a[5]=(bf16_t)f1.y; a[6]=(bf16_t)f1.z; a[7]=(bf16_t)f1.w;
            a[8]=(bf16_t)f2.x; a[9]=(bf16_t)f2.y; a[10]=(bf16_t)f2.z; a[11]=(bf16_t)f2.w;
            a[12]=(bf16_t)f3.x; a[13]=(bf16_t)f3.y; a[14]=(bf16_t)f3.z; a[15]=(bf16_t)f3.w;
            afr[s] = a;
        }

        float part[8] = {0.f,0.f,0.f,0.f,0.f,0.f,0.f,0.f};
        for (int n = 0; n < 16; ++n) {
            v8f acc = {0.f,0.f,0.f,0.f,0.f,0.f,0.f,0.f};
            const int e = 16 * n + ml;                    // output column for B frag
            const bf16_t* wrow = wT + e * WROW;
            #pragma unroll
            for (int s = 0; s < 8; ++s) {
                // B[d][e] = W_pr1[e][d]: 16 consecutive d's, 16B aligned in LDS
                v16bf bfr;
                const unsigned int* bp = (const unsigned int*)(wrow + 32 * s + 16 * half);
                unsigned int* dst = (unsigned int*)&bfr;
                #pragma unroll
                for (int i = 0; i < 8; ++i) dst[i] = bp[i];
                acc = __builtin_amdgcn_wmma_f32_16x16x32_bf16(
                        false, afr[s], false, bfr, (short)0, acc, false, false);
            }
            const float bb = sm[SM_BPR1 + e];
            const float ww = sm[SM_WPR2 + e];
            #pragma unroll
            for (int j = 0; j < 8; ++j) {
                float h = acc[j] + bb;          // + b_pr1
                h = fmaxf(h, 0.f);              // ReLU
                part[j] = fmaf(h, ww, part[j]); // * w_pr2, accumulate over e
            }
        }
        // reduce across the 16 lanes of each half (same M rows, different e)
        #pragma unroll
        for (int off = 1; off < 16; off <<= 1) {
            #pragma unroll
            for (int j = 0; j < 8; ++j) part[j] += __shfl_xor(part[j], off);
        }
        if (ml == 0) {
            const float b2 = b_pr2[0];
            #pragma unroll
            for (int j = 0; j < 8; ++j)   // row m = 16w + 8*half + j
                sm[SM_MPR + 16 * wv + 8 * half + j] = part[j] + b2 - 0.5f; // - sigma^2/2
        }
    }
    __syncthreads();

    // ---------- phase D: KL + reparameterized sample + softmax ----------
    float term = 0.f, samp = -3.4e38f;
    if (t < NKEY) {
        const float lv  = q_lv + sm[SM_DLV + t] + b_lv[0];
        const float sp  = expf(0.5f * lv);
        const float sp2 = sp * sp;
        const float mp  = q_mu + sm[SM_DMU + t] + b_mu[0] - 0.5f * sp2;
        const float df  = mp - sm[SM_MPR + t];
        term = -0.5f * lv + 0.5f * (sp2 + df * df) - 0.5f;
        samp = expf(mp + sp * eps[(size_t)q * NKEY + t]);   // TEMP == 1
    }
    const float klq = blockSum(red, t, term);
    if (t == 0) klPart[q] = klq;

    const float mx   = blockMax(red, t, samp);
    const float ex   = (t < NKEY) ? expf(samp - mx) : 0.f;
    const float ssum = blockSum(red, t, ex);
    if (t < NKEY) sm[SM_WTS + t] = ex / ssum;
    __syncthreads();

    // ---------- phase E: context = weights @ V, then LayerNorm ----------
    const float* Vq = V + (size_t)q * (NKEY * DIM);
    float ctx = 0.f;
    #pragma unroll 4
    for (int k = 0; k < NKEY; ++k)
        ctx = fmaf(sm[SM_WTS + k], Vq[k * DIM + t], ctx);

    const float mean = blockSum(red, t, ctx) * (1.f / 256.f);
    const float dv   = ctx - mean;
    const float var  = blockSum(red, t, dv * dv) * (1.f / 256.f);
    out[(size_t)q * DIM + t] =
        dv * (1.0f / sqrtf(var + 1e-5f)) * sm[SM_GAM + t] + sm[SM_BET + t];
}

// ---------- kernel 3: deterministic KL reduction (fixed order, no fp atomics) ----------
__global__ __launch_bounds__(256) void klReduce_kernel(const float* __restrict__ part,
                                                       float* __restrict__ outKL) {
    __shared__ float red[256];
    int t = threadIdx.x;
    float s = 0.f;
    #pragma unroll
    for (int i = 0; i < 16; ++i) s += part[t + i * 256];
    red[t] = s; __syncthreads();
    #pragma unroll
    for (int off = 128; off > 0; off >>= 1) {
        if (t < off) red[t] += red[t + off];
        __syncthreads();
    }
    if (t == 0) outKL[0] = red[0] * (1.0f / ((float)NQ * (float)NKEY));
}

extern "C" void kernel_launch(void* const* d_in, const int* in_sizes, int n_in,
                              void* d_out, int out_size, void* d_ws, size_t ws_size,
                              hipStream_t stream) {
    (void)in_sizes; (void)n_in; (void)out_size; (void)ws_size;
    const float* Q      = (const float*)d_in[0];
    const float* K      = (const float*)d_in[1];
    const float* V      = (const float*)d_in[2];
    const float* eps    = (const float*)d_in[3];
    const float* w_mu_q = (const float*)d_in[4];
    const float* w_mu_k = (const float*)d_in[5];
    const float* b_mu   = (const float*)d_in[6];
    const float* w_lv_q = (const float*)d_in[7];
    const float* w_lv_k = (const float*)d_in[8];
    const float* b_lv   = (const float*)d_in[9];
    const float* W_pr1  = (const float*)d_in[10];
    const float* b_pr1  = (const float*)d_in[11];
    const float* w_pr2  = (const float*)d_in[12];
    const float* b_pr2  = (const float*)d_in[13];
    const float* gamma  = (const float*)d_in[14];
    const float* beta   = (const float*)d_in[15];

    float*  out    = (float*)d_out;                       // context [4096*256], then kl
    bf16_t* Wbf    = (bf16_t*)d_ws;                       // 131072 B
    float*  klPart = (float*)((char*)d_ws + 131072);      // 4096 floats

    convW_kernel<<<256, 256, 0, stream>>>(W_pr1, Wbf);
    vattn_main<<<NQ, 256, SMEM_BYTES, stream>>>(
        Q, K, V, eps, w_mu_q, w_mu_k, b_mu, w_lv_q, w_lv_k, b_lv,
        b_pr1, w_pr2, b_pr2, gamma, beta, Wbf, out, klPart);
    klReduce_kernel<<<1, 256, 0, stream>>>(klPart, out + (size_t)NQ * DIM);
}
